// GATv2_22213570854984
// MI455X (gfx1250) — compile-verified, matrix operand order checked by reading, not run on hardware
//
#include <hip/hip_runtime.h>
#include <hip/hip_bf16.h>
#include <cmath>

// ---------------------------------------------------------------------------
// Types for CDNA5 WMMA (wave32, 16x16x32 f16 -> f32 accumulate)
// ---------------------------------------------------------------------------
typedef __attribute__((ext_vector_type(16))) _Float16 v16h;
typedef __attribute__((ext_vector_type(8)))  float    v8f;

#define NEG_SLOPE 0.2f
#define BN_EPS    1e-5f
#define HEADS     4

// ---------------------------------------------------------------------------
// Helpers
// ---------------------------------------------------------------------------
__device__ __forceinline__ void edge_pair(const int* __restrict__ ei, int E,
                                          long long e, int& s, int& d) {
  if (e < (long long)E) { s = ei[e]; d = ei[(long long)E + e]; }
  else { int i = (int)(e - E); s = i; d = i; }   // self loops appended after edges
}

// float atomic max via sign-split int/uint atomics (lowers to
// global_atomic_max_i32 / global_atomic_min_u32). Init value must be -inf.
__device__ __forceinline__ void atomicMaxFloat(float* addr, float val) {
  if (val >= 0.0f) atomicMax((int*)addr, __float_as_int(val));
  else             atomicMin((unsigned int*)addr, (unsigned int)__float_as_int(val));
}

__global__ void fill_kernel(float* __restrict__ p, float v, long long n) {
  long long i = (long long)blockIdx.x * blockDim.x + threadIdx.x;
  if (i < n) p[i] = v;
}

// ---------------------------------------------------------------------------
// WMMA GEMM: Y[M,NC] = A[M,K] @ W[K,NC] + bias[NC]   (f32 in/out, f16 compute)
//
// W is staged into LDS *pre-packed in WMMA B-fragment order*: for fragment
// f = kt*NT + t, lane L, the 16 halves of lane L's B operand live contiguously
// at sW[(f*32 + L)*16 .. +15].  The inner loop then loads each fragment with
// one v16h deref (=> 2x ds_load_b128) instead of 16 scalar ds_load_u16.
//
// A fragment (CDNA5 16-bit A 16x32 layout): lane m = lane&15 is the row,
// half = lane>>4 selects the +8 K offset; per-lane data is two contiguous
// 8-float runs of the row => four float4 global loads + packed f16 converts.
//
// K multiple of 32; NC in {32, 64}. One wave owns a 16 x NC strip.
// ---------------------------------------------------------------------------
template <int NC>
__global__ void wmma_gemm_bias(const float* __restrict__ A,
                               const float* __restrict__ W,
                               const float* __restrict__ bias,
                               float* __restrict__ Y, int M, int K) {
  constexpr int NT = NC / 16;
  extern __shared__ _Float16 sW[];   // K*NC halves, fragment-packed

  const int KT = K / 32;

  // Cooperative stage + permute of W into fragment order (one-time per block).
  // i enumerates (frag f, lane, v) triples; each handles a packed K-pair.
  for (int i = threadIdx.x; i < KT * NT * 32 * 8; i += blockDim.x) {
    const int v     = i & 7;
    const int lane2 = (i >> 3) & 31;
    const int f     = i >> 8;       // f = kt*NT + t
    const int t     = f % NT;
    const int kt    = f / NT;
    const int col   = t * 16 + (lane2 & 15);
    const int half2 = lane2 >> 4;
    const int kk    = kt * 32 + ((v < 4) ? (v * 2 + half2 * 8)
                                         : (16 + (v - 4) * 2 + half2 * 8));
    sW[i * 2 + 0] = (_Float16)W[kk * NC + col];
    sW[i * 2 + 1] = (_Float16)W[(kk + 1) * NC + col];
  }
  __syncthreads();

  const int lane = threadIdx.x & 31;
  const int wave = threadIdx.x >> 5;
  const int m    = lane & 15;
  const int half = lane >> 4;
  const long long row0 = ((long long)blockIdx.x * (blockDim.x >> 5) + wave) * 16;
  const long long row  = row0 + m;
  const long long arow = (row < M) ? row : (long long)(M - 1);  // clamp, masked on store
  const float* __restrict__ Arow = A + arow * (long long)K;

  v8f acc[NT];
#pragma unroll
  for (int t = 0; t < NT; ++t) acc[t] = (v8f)(0.0f);

  for (int kt = 0; kt < KT; ++kt) {
    const int k0 = kt * 32;
    // ---- A fragment: two contiguous 8-float runs -> 4x global_load_b128 ----
    const float4 a01 = *(const float4*)(Arow + k0 + half * 8);
    const float4 a23 = *(const float4*)(Arow + k0 + half * 8 + 4);
    const float4 a45 = *(const float4*)(Arow + k0 + 16 + half * 8);
    const float4 a67 = *(const float4*)(Arow + k0 + 16 + half * 8 + 4);
    v16h a;
    a[0]  = (_Float16)a01.x; a[1]  = (_Float16)a01.y;
    a[2]  = (_Float16)a01.z; a[3]  = (_Float16)a01.w;
    a[4]  = (_Float16)a23.x; a[5]  = (_Float16)a23.y;
    a[6]  = (_Float16)a23.z; a[7]  = (_Float16)a23.w;
    a[8]  = (_Float16)a45.x; a[9]  = (_Float16)a45.y;
    a[10] = (_Float16)a45.z; a[11] = (_Float16)a45.w;
    a[12] = (_Float16)a67.x; a[13] = (_Float16)a67.y;
    a[14] = (_Float16)a67.z; a[15] = (_Float16)a67.w;

    // ---- B fragments: contiguous 32B per lane -> 2x ds_load_b128 each ----
#pragma unroll
    for (int t = 0; t < NT; ++t) {
      const _Float16* bp = sW + (size_t)((kt * NT + t) * 32 + lane) * 16;
      const v16h b = *(const v16h*)bp;
      acc[t] = __builtin_amdgcn_wmma_f32_16x16x32_f16(
          /*neg_a=*/false, a, /*neg_b=*/false, b,
          /*c_mod=*/(short)0, acc[t], /*reuse_a=*/false, /*reuse_b=*/false);
    }
  }

  // C/D layout: VGPR r -> row (r + half*8), col = t*16 + (lane&15)
#pragma unroll
  for (int t = 0; t < NT; ++t) {
    const int col = t * 16 + m;
    const float bv = bias[col];
#pragma unroll
    for (int r = 0; r < 8; ++r) {
      const long long orow = row0 + r + half * 8;
      if (orow < M) Y[orow * NC + col] = acc[t][r] + bv;
    }
  }
}

// ---------------------------------------------------------------------------
// Edge pass 1: score[e,h] = att . leaky_relu(xl[src] + xr[dst]); segment max.
// GROUP = HEADS*C threads per edge (coalesced channel reads).
// ---------------------------------------------------------------------------
template <int GROUP, int C>
__global__ void edge_score_kernel(const float* __restrict__ xl,
                                  const float* __restrict__ xr,
                                  const float* __restrict__ att,
                                  const int* __restrict__ ei, int E, int Etot,
                                  float* __restrict__ score,
                                  float* __restrict__ smax) {
  long long gid = (long long)blockIdx.x * blockDim.x + threadIdx.x;
  long long e = gid / GROUP;
  int j = (int)(gid % GROUP);
  if (e >= (long long)Etot) return;
  int s, d;
  edge_pair(ei, E, e, s, d);
  float v = xl[(long long)s * GROUP + j] + xr[(long long)d * GROUP + j];
  v = (v > 0.0f) ? v : NEG_SLOPE * v;
  v *= att[j];
#pragma unroll
  for (int o = C / 2; o > 0; o >>= 1) v += __shfl_xor(v, o, 32);
  if ((j % C) == 0) {
    const int h = j / C;
    score[e * HEADS + h] = v;
    atomicMaxFloat(&smax[(long long)d * HEADS + h], v);
  }
}

// ---------------------------------------------------------------------------
// Edge pass 2: w = exp(score - smax[dst]); denom[dst,h] += w (in-place on score)
// ---------------------------------------------------------------------------
__global__ void edge_weight_kernel(const int* __restrict__ ei, int E, int Etot,
                                   float* __restrict__ score,
                                   const float* __restrict__ smax,
                                   float* __restrict__ denom) {
  long long gid = (long long)blockIdx.x * blockDim.x + threadIdx.x;
  long long e = gid / HEADS;
  int h = (int)(gid % HEADS);
  if (e >= (long long)Etot) return;
  int s, d;
  edge_pair(ei, E, e, s, d);
  const float w = __expf(score[e * HEADS + h] - smax[(long long)d * HEADS + h]);
  score[e * HEADS + h] = w;
  atomicAdd(&denom[(long long)d * HEADS + h], w);
}

// ---------------------------------------------------------------------------
// Edge pass 3: out[dst] += (w/denom[dst]) * xl[src]
// ---------------------------------------------------------------------------
template <int GROUP, int C>
__global__ void edge_aggr_kernel(const float* __restrict__ xl,
                                 const int* __restrict__ ei, int E, int Etot,
                                 const float* __restrict__ score,
                                 const float* __restrict__ denom,
                                 float* __restrict__ agg) {
  long long gid = (long long)blockIdx.x * blockDim.x + threadIdx.x;
  long long e = gid / GROUP;
  int j = (int)(gid % GROUP);
  if (e >= (long long)Etot) return;
  int s, d;
  edge_pair(ei, E, e, s, d);
  const int h = j / C;
  const float alpha = score[e * HEADS + h] / denom[(long long)d * HEADS + h];
  atomicAdd(&agg[(long long)d * GROUP + j], alpha * xl[(long long)s * GROUP + j]);
}

// ---------------------------------------------------------------------------
// Fused: h += bias; h = ELU(h); accumulate per-channel sum/sumsq (LDS reduce).
// Optionally mirrors the ELU output to `save` (layer-4 out4).
// ---------------------------------------------------------------------------
__global__ void bias_elu_stats_kernel(float* __restrict__ h,
                                      const float* __restrict__ bias,
                                      float* __restrict__ stats, int Nn, int D,
                                      float* __restrict__ save) {
  __shared__ float ssum[64];
  __shared__ float ssq[64];
  const int tid = threadIdx.x;
  if (tid < D) { ssum[tid] = 0.0f; ssq[tid] = 0.0f; }
  __syncthreads();
  const long long gid = (long long)blockIdx.x * blockDim.x + tid;
  const long long total = (long long)Nn * D;
  if (gid < total) {
    const int ch = (int)(gid % D);
    float v = h[gid] + bias[ch];
    v = (v > 0.0f) ? v : (__expf(v) - 1.0f);
    h[gid] = v;
    if (save) save[gid] = v;
    atomicAdd(&ssum[ch], v);
    atomicAdd(&ssq[ch], v * v);
  }
  __syncthreads();
  if (tid < D) {
    atomicAdd(&stats[tid], ssum[tid]);
    atomicAdd(&stats[D + tid], ssq[tid]);
  }
}

__global__ void bn_apply_kernel(const float* __restrict__ h,
                                const float* __restrict__ stats,
                                const float* __restrict__ gamma,
                                const float* __restrict__ beta,
                                float* __restrict__ out, int Nn, int D) {
  const long long gid = (long long)blockIdx.x * blockDim.x + threadIdx.x;
  if (gid >= (long long)Nn * D) return;
  const int ch = (int)(gid % D);
  const float inv = 1.0f / (float)Nn;
  const float mean = stats[ch] * inv;
  const float var = stats[D + ch] * inv - mean * mean;
  out[gid] = (h[gid] - mean) * rsqrtf(var + BN_EPS) * gamma[ch] + beta[ch];
}

// Classifier: logits[N,2] = bn4[N,32] @ W[32,2] + b[2]  (too skinny for WMMA)
__global__ void classify_kernel(const float* __restrict__ bn4,
                                const float* __restrict__ Wc,
                                const float* __restrict__ bc,
                                float* __restrict__ logits, int Nn) {
  const int row = blockIdx.x * blockDim.x + threadIdx.x;
  if (row >= Nn) return;
  float a0 = bc[0], a1 = bc[1];
#pragma unroll
  for (int k = 0; k < 32; ++k) {
    const float v = bn4[(long long)row * 32 + k];
    a0 += v * Wc[k * 2 + 0];
    a1 += v * Wc[k * 2 + 1];
  }
  logits[(long long)row * 2 + 0] = a0;
  logits[(long long)row * 2 + 1] = a1;
}

// ---------------------------------------------------------------------------
// Host orchestration
// ---------------------------------------------------------------------------
struct ConvP { const float *Wl, *bl, *Wr, *br, *att, *b; };

static inline int cdiv_ll(long long a, long long b) { return (int)((a + b - 1) / b); }

extern "C" void kernel_launch(void* const* d_in, const int* in_sizes, int n_in,
                              void* d_out, int out_size, void* d_ws, size_t ws_size,
                              hipStream_t stream) {
  const int GENE = 256, HID = 64, HID2 = 32;
  const int N = in_sizes[0] / GENE;
  const int E = in_sizes[1] / 2;
  const int Etot = E + N;  // add_self_loops

  const float* x  = (const float*)d_in[0];
  const int*   ei = (const int*)d_in[1];

  ConvP cv[4];
  int idx = 2;
  for (int l = 0; l < 4; ++l) {
    cv[l].Wl  = (const float*)d_in[idx + 0];
    cv[l].bl  = (const float*)d_in[idx + 1];
    cv[l].Wr  = (const float*)d_in[idx + 2];
    cv[l].br  = (const float*)d_in[idx + 3];
    cv[l].att = (const float*)d_in[idx + 4];
    cv[l].b   = (const float*)d_in[idx + 5];
    idx += 6;
  }
  const float* bn_g[4]; const float* bn_b[4];
  for (int l = 0; l < 4; ++l) { bn_g[l] = (const float*)d_in[idx]; bn_b[l] = (const float*)d_in[idx + 1]; idx += 2; }
  const float* cls_W = (const float*)d_in[idx];
  const float* cls_b = (const float*)d_in[idx + 1];

  // Output layout: logits[N,2] | out4[N,32] | bn4_out[N,32]
  float* logits = (float*)d_out;
  float* out4   = logits + (long long)N * 2;
  float* bn4    = out4 + (long long)N * 32;

  // Workspace carve-out (floats)
  float* ws = (float*)d_ws;
  long long off = 0;
  auto alloc = [&](long long n) { float* p = ws + off; off += n; return p; };
  float* xl    = alloc((long long)N * HID);
  float* xr    = alloc((long long)N * HID);
  float* bufA  = alloc((long long)N * HID);
  float* agg   = alloc((long long)N * HID);
  float* score = alloc((long long)Etot * HEADS);
  float* smax  = alloc((long long)N * HEADS);
  float* denom = alloc((long long)N * HEADS);
  float* stats = alloc(128);

  const int TB = 256;

  for (int l = 0; l < 4; ++l) {
    const float* cur = (l == 0) ? x : bufA;
    const int Din  = (l == 0) ? GENE : HID;
    const int Dout = (l == 3) ? HID2 : HID;

    // --- dense transforms via WMMA (8 waves * 16 rows = 128 rows per block)
    const int gblk = cdiv_ll(N, 128);
    const size_t shbytes = (size_t)Din * Dout * 2;  // f16 fragment-packed weights
    if (Dout == 64) {
      wmma_gemm_bias<64><<<gblk, TB, shbytes, stream>>>(cur, cv[l].Wl, cv[l].bl, xl, N, Din);
      wmma_gemm_bias<64><<<gblk, TB, shbytes, stream>>>(cur, cv[l].Wr, cv[l].br, xr, N, Din);
    } else {
      wmma_gemm_bias<32><<<gblk, TB, shbytes, stream>>>(cur, cv[l].Wl, cv[l].bl, xl, N, Din);
      wmma_gemm_bias<32><<<gblk, TB, shbytes, stream>>>(cur, cv[l].Wr, cv[l].br, xr, N, Din);
    }

    // --- init segment buffers
    fill_kernel<<<cdiv_ll((long long)N * HEADS, TB), TB, 0, stream>>>(smax, -INFINITY, (long long)N * HEADS);
    fill_kernel<<<cdiv_ll((long long)N * HEADS, TB), TB, 0, stream>>>(denom, 0.0f, (long long)N * HEADS);
    fill_kernel<<<cdiv_ll((long long)N * Dout, TB), TB, 0, stream>>>(agg, 0.0f, (long long)N * Dout);
    fill_kernel<<<1, 128, 0, stream>>>(stats, 0.0f, 128);

    // --- edge phase
    const long long tEdgeCh = (long long)Etot * Dout;
    if (Dout == 64) {
      edge_score_kernel<64, 16><<<cdiv_ll(tEdgeCh, TB), TB, 0, stream>>>(xl, xr, cv[l].att, ei, E, Etot, score, smax);
    } else {
      edge_score_kernel<32, 8><<<cdiv_ll(tEdgeCh, TB), TB, 0, stream>>>(xl, xr, cv[l].att, ei, E, Etot, score, smax);
    }
    edge_weight_kernel<<<cdiv_ll((long long)Etot * HEADS, TB), TB, 0, stream>>>(ei, E, Etot, score, smax, denom);
    if (Dout == 64) {
      edge_aggr_kernel<64, 16><<<cdiv_ll(tEdgeCh, TB), TB, 0, stream>>>(xl, ei, E, Etot, score, denom, agg);
    } else {
      edge_aggr_kernel<32, 8><<<cdiv_ll(tEdgeCh, TB), TB, 0, stream>>>(xl, ei, E, Etot, score, denom, agg);
    }

    // --- node phase: bias + ELU + BN
    const long long tNode = (long long)N * Dout;
    bias_elu_stats_kernel<<<cdiv_ll(tNode, TB), TB, 0, stream>>>(
        agg, cv[l].b, stats, N, Dout, (l == 3) ? out4 : nullptr);
    bn_apply_kernel<<<cdiv_ll(tNode, TB), TB, 0, stream>>>(
        agg, stats, bn_g[l], bn_b[l], (l == 3) ? bn4 : bufA, N, Dout);
  }

  classify_kernel<<<cdiv_ll(N, TB), TB, 0, stream>>>(bn4, cls_W, cls_b, logits, N);
}
